// ScaledDotProduct_38646115729876
// MI455X (gfx1250) — matrix operand order repaired and checked by reading, not compiled, not run on hardware
//
#include <hip/hip_runtime.h>

typedef __attribute__((ext_vector_type(16))) _Float16 v16h;
typedef __attribute__((ext_vector_type(8)))  _Float16 v8h;
typedef __attribute__((ext_vector_type(2)))  _Float16 v2h;
typedef __attribute__((ext_vector_type(8)))  float    v8f;
typedef __attribute__((ext_vector_type(4)))  float    v4f;

#define SEQ 2048
#define DH  64
#define LOG2E 1.44269504088896340736f
#define NEG2  (-1000000000.0f * LOG2E)     // mask constant in base-2 domain
#define QSCALE (0.125f * LOG2E)            // 1/sqrt(64) * log2(e), folded into Q

// xor-16 lane exchange via ds_swizzle (group-of-32: and=0x1F, or=0, xor=16)
__device__ __forceinline__ float swzf(float x) {
    return __int_as_float(__builtin_amdgcn_ds_swizzle(__float_as_int(x), 0x401F));
}
__device__ __forceinline__ int swzi(int x) {
    return __builtin_amdgcn_ds_swizzle(x, 0x401F);
}
__device__ __forceinline__ int pkh2(float a, float b) {   // pack two f32 -> {f16,f16} dword
    union { v2h h; int i; } u;
    u.h[0] = (_Float16)a; u.h[1] = (_Float16)b;
    return u.i;
}

__global__ __launch_bounds__(256) void fa_causal_kernel(
    const float* __restrict__ Q, const float* __restrict__ K,
    const float* __restrict__ V, const float* __restrict__ AM,
    float* __restrict__ O)
{
    // Double-buffered LDS tiles:
    //  sK : K tile row-major f16 [32 keys][64 d]
    //  sVt: V tile transposed f16 [64 d][stride 40 keys] (16B-aligned fragment rows)
    //  sAM: per-key additive mask (base-2 domain)
    __shared__ __align__(16) _Float16 sK [2][32 * 64];
    __shared__ __align__(16) _Float16 sVt[2][64 * 40];
    __shared__ __align__(16) float    sAM[2][32];

    const int tid  = threadIdx.x;
    const int lane = tid & 31;
    const int wv   = __builtin_amdgcn_readfirstlane(tid >> 5);   // scalar wave id
    const int l15  = lane & 15;
    const int lhi  = lane >> 4;
    const int bh   = blockIdx.y;
    const int b    = bh >> 4;                 // H = 16
    const int qblk = blockIdx.x;
    const int q0   = qblk * 128 + wv * 16;    // scalar

    const float* qp  = Q + (size_t)bh * SEQ * DH;
    const float* kp  = K + (size_t)bh * SEQ * DH;
    const float* vp  = V + (size_t)bh * SEQ * DH;
    const float* amp = AM + (size_t)b * SEQ;
    float*       op  = O + (size_t)bh * SEQ * DH;

    // ---- Q^T B-fragments, pre-scaled by 1/sqrt(d)*log2e ----
    v16h qb0, qb1;
    {
        const float* qrow = qp + (size_t)(q0 + l15) * DH + lhi * 8;
        #pragma unroll
        for (int c = 0; c < 2; ++c) {
            const v4f f0 = *(const v4f*)(qrow + c * 32 + 0);
            const v4f f1 = *(const v4f*)(qrow + c * 32 + 4);
            const v4f f2 = *(const v4f*)(qrow + c * 32 + 16);
            const v4f f3 = *(const v4f*)(qrow + c * 32 + 20);
            v16h dst;
            #pragma unroll
            for (int i = 0; i < 4; ++i) {
                dst[i]      = (_Float16)(f0[i] * QSCALE);
                dst[4 + i]  = (_Float16)(f1[i] * QSCALE);
                dst[8 + i]  = (_Float16)(f2[i] * QSCALE);
                dst[12 + i] = (_Float16)(f3[i] * QSCALE);
            }
            if (c == 0) qb0 = dst; else qb1 = dst;
        }
    }

    v8f o0 = {}, o1 = {}, o2 = {}, o3 = {};
    float m_i = -__builtin_inff();
    float l_i = 0.0f;

    const int nkt = qblk * 4 + 4;

    // ---- software pipeline: registers hold the *current* tile's global f32 data ----
    const int row = tid >> 3;        // 0..31
    const int c8  = (tid & 7) * 8;   // 0..56
    const float* kgp = kp + (size_t)row * DH + c8;
    const float* vgp = vp + (size_t)row * DH + c8;
    v4f kr0 = *(const v4f*)kgp,       kr1 = *(const v4f*)(kgp + 4);
    v4f vr0 = *(const v4f*)vgp,       vr1 = *(const v4f*)(vgp + 4);
    float amreg = (tid < 32) ? amp[tid] : 0.0f;

    for (int kt = 0; kt < nkt; ++kt) {
        const int kbase = kt * 32;    // scalar
        const int buf   = kt & 1;     // scalar double-buffer select
        _Float16* sKb  = sK [buf];
        _Float16* sVb  = sVt[buf];
        float*    sAMb = sAM[buf];

        // ---- stage current tile regs -> LDS (f16), then launch next tile's loads ----
        {
            v8h kh;
            #pragma unroll
            for (int i = 0; i < 4; ++i) { kh[i] = (_Float16)kr0[i]; kh[4 + i] = (_Float16)kr1[i]; }
            *(v8h*)(&sKb[row * 64 + c8]) = kh;
            #pragma unroll
            for (int i = 0; i < 4; ++i) sVb[(c8 + i) * 40 + row]     = (_Float16)vr0[i];
            #pragma unroll
            for (int i = 0; i < 4; ++i) sVb[(c8 + 4 + i) * 40 + row] = (_Float16)vr1[i];
            if (tid < 32) sAMb[tid] = (1.0f - amreg) * NEG2;

            if (kt + 1 < nkt) {      // in flight across the compute phase below
                kgp += 32 * DH; vgp += 32 * DH;
                kr0 = *(const v4f*)kgp; kr1 = *(const v4f*)(kgp + 4);
                vr0 = *(const v4f*)vgp; vr1 = *(const v4f*)(vgp + 4);
                if (tid < 32) amreg = amp[kbase + 32 + tid];
            }
        }
        __syncthreads();   // single barrier per tile: stage(kt) complete; buffer kt&1 was
                           // last read in compute(kt-2), proven finished by barrier(kt-1)

        if (kbase <= q0 + 15) {      // scalar causal tile skip
            // ---- S^T = K * Q^T ----
            v8f st0 = {}, st1 = {};
            #pragma unroll
            for (int tk = 0; tk < 2; ++tk) {
                #pragma unroll
                for (int c = 0; c < 2; ++c) {
                    const _Float16* ap = &sKb[(tk * 16 + l15) * 64 + c * 32 + lhi * 8];
                    const v8h lo = *(const v8h*)ap;
                    const v8h hi = *(const v8h*)(ap + 16);
                    v16h af;
                    #pragma unroll
                    for (int e = 0; e < 8; ++e) { af[e] = lo[e]; af[8 + e] = hi[e]; }
                    if (tk == 0)
                        st0 = __builtin_amdgcn_wmma_f32_16x16x32_f16(false, af, false, (c ? qb1 : qb0), (short)0, st0, false, false);
                    else
                        st1 = __builtin_amdgcn_wmma_f32_16x16x32_f16(false, af, false, (c ? qb1 : qb0), (short)0, st1, false, false);
                }
            }

            // ---- base-2 online softmax (per query = per lane) ----
            const int qrow = q0 + l15;
            float p0[8], p1[8], corr;
            auto softmax = [&](bool domask) {
                float sc0[8], sc1[8];
                float tmax = -__builtin_inff();
                #pragma unroll
                for (int tk = 0; tk < 2; ++tk) {
                    const v4f A0 = *(const v4f*)&sAMb[tk * 16 + lhi * 8];
                    const v4f A1 = *(const v4f*)&sAMb[tk * 16 + lhi * 8 + 4];
                    #pragma unroll
                    for (int r = 0; r < 8; ++r) {
                        float s = (tk ? st1[r] : st0[r]) + ((r < 4) ? A0[r] : A1[r - 4]);
                        if (domask) {
                            const int key = kbase + tk * 16 + lhi * 8 + r;
                            if (key > qrow) s = NEG2;
                        }
                        (tk ? sc1 : sc0)[r] = s;
                        tmax = fmaxf(tmax, s);
                    }
                }
                tmax = fmaxf(tmax, swzf(tmax));
                const float m_new = fmaxf(m_i, tmax);
                corr = __builtin_amdgcn_exp2f(m_i - m_new);
                float rs = 0.0f;
                #pragma unroll
                for (int r = 0; r < 8; ++r) { p0[r] = __builtin_amdgcn_exp2f(sc0[r] - m_new); rs += p0[r]; }
                #pragma unroll
                for (int r = 0; r < 8; ++r) { p1[r] = __builtin_amdgcn_exp2f(sc1[r] - m_new); rs += p1[r]; }
                rs += swzf(rs);
                l_i = l_i * corr + rs;
                m_i = m_new;
            };
            if (kbase + 31 <= q0) softmax(false);   // interior tile: no causal compares
            else                  softmax(true);    // diagonal tile

            o0 *= corr; o1 *= corr; o2 *= corr; o3 *= corr;

            // ---- P^T B-fragment via packed xor-16 exchanges ----
            union { v16h h; int i[8]; } pbu;
            #pragma unroll
            for (int v2 = 0; v2 < 4; ++v2) {
                const int a  = pkh2(p0[2 * v2], p0[2 * v2 + 1]);
                const int bq = pkh2(p1[2 * v2], p1[2 * v2 + 1]);
                const int sa = swzi(a);
                const int sb = swzi(bq);
                pbu.i[v2]     = lhi ? sb : a;    // keys (lhi?16:0) + 2*v2..
                pbu.i[4 + v2] = lhi ? bq : sa;   // keys (lhi?24:8) + 2*v2..
            }
            const v16h pb = pbu.h;

            // ---- O^T += V^T * P^T ----
            #pragma unroll
            for (int dt = 0; dt < 4; ++dt) {
                const _Float16* vap = &sVb[(dt * 16 + l15) * 40 + lhi * 16];
                const v8h lo = *(const v8h*)vap;
                const v8h hi = *(const v8h*)(vap + 8);
                v16h vf;
                #pragma unroll
                for (int e = 0; e < 8; ++e) { vf[e] = lo[e]; vf[8 + e] = hi[e]; }
                if      (dt == 0) o0 = __builtin_amdgcn_wmma_f32_16x16x32_f16(false, vf, false, pb, (short)0, o0, false, false);
                else if (dt == 1) o1 = __builtin_amdgcn_wmma_f32_16x16x32_f16(false, vf, false, pb, (short)0, o1, false, false);
                else if (dt == 2) o2 = __builtin_amdgcn_wmma_f32_16x16x32_f16(false, vf, false, pb, (short)0, o2, false, false);
                else              o3 = __builtin_amdgcn_wmma_f32_16x16x32_f16(false, vf, false, pb, (short)0, o3, false, false);
            }
        }
    }

    // ---- epilogue ----
    const float inv_l = 1.0f / l_i;
    float* obase = op + (size_t)(q0 + l15) * DH + lhi * 8;   // d = dt*16 + lhi*8 + r
    #pragma unroll
    for (int dt = 0; dt < 4; ++dt) {
        const v8f oc = (dt == 0) ? o0 : (dt == 1) ? o1 : (dt == 2) ? o2 : o3;
        v4f x, y;
        #pragma unroll
        for (int i = 0; i < 4; ++i) { x[i] = oc[i] * inv_l; y[i] = oc[4 + i] * inv_l; }
        *(v4f*)(obase + dt * 16)     = x;
        *(v4f*)(obase + dt * 16 + 4) = y;
    }
}

extern "C" void kernel_launch(void* const* d_in, const int* in_sizes, int n_in,
                              void* d_out, int out_size, void* d_ws, size_t ws_size,
                              hipStream_t stream) {
    const float* q  = (const float*)d_in[0];
    const float* k  = (const float*)d_in[1];
    const float* v  = (const float*)d_in[2];
    const float* am = (const float*)d_in[3];
    float* out = (float*)d_out;
    (void)in_sizes; (void)n_in; (void)out_size; (void)d_ws; (void)ws_size;

    dim3 grid(SEQ / 128, 2 * 16);   // (query blocks of 128, B*H)
    dim3 block(256);                // 8 wave32s, one 16-row query tile each
    fa_causal_kernel<<<grid, block, 0, stream>>>(q, k, v, am, out);
}
